// MultiheadAttention_75170517615180
// MI455X (gfx1250) — matrix operand order repaired and checked by reading, not compiled, not run on hardware
//
#include <hip/hip_runtime.h>
#include <hip/hip_bf16.h>
#include <math.h>

typedef __bf16 bf16_t;
typedef __attribute__((ext_vector_type(16))) __bf16 v16bf;
typedef __attribute__((ext_vector_type(8)))  __bf16 v8bf;
typedef __attribute__((ext_vector_type(4)))  __bf16 v4bf;
typedef __attribute__((ext_vector_type(8)))  float  v8f;
typedef __attribute__((ext_vector_type(4)))  float  v4f;

#define D_MODEL 1024
#define N_HEADS 16
#define D_KH    64
#define B_SZ    8
#define S_LEN   1024

// ---------------------------------------------------------------------------
// fp32 -> bf16 conversion, 4 elements / thread. n must be a multiple of 1024.
// ---------------------------------------------------------------------------
__global__ __launch_bounds__(256)
void cvt_f32_bf16_kernel(const float* __restrict__ src, bf16_t* __restrict__ dst)
{
    const size_t i = (size_t)blockIdx.x * 256 + threadIdx.x;
    v4f v = ((const v4f*)src)[i];
    v4bf o;
#pragma unroll
    for (int c = 0; c < 4; ++c) o[c] = (bf16_t)v[c];
    ((v4bf*)dst)[i] = o;
}

// ---------------------------------------------------------------------------
// Batched WMMA GEMM:  C[m,n] = alpha * ( sum_k A[m,k]*B[n,k] + bias[n] )
//   A,B bf16 row-major (torch-Linear weight orientation for B: W[n][k]).
//   C: f32 (non-temporal store; big streaming outputs) or bf16 (cache-resident
//      intermediates); TRANSC stores C[n*ldC + m].
//   per-z offsets: base + (z/HH)*outer + (z%HH)*inner   (elements)
// Wave tile 64x32 (4x2 subtiles), WG = WM_WAVES x WN_WAVES waves.
// ---------------------------------------------------------------------------
template<typename CT, bool TRANSC, int WM_WAVES, int WN_WAVES>
__global__ __launch_bounds__(WM_WAVES * WN_WAVES * 32)
void wmma_gemm_kernel(const bf16_t* __restrict__ A, long long ldA, long long aOuter, long long aInner,
                      const bf16_t* __restrict__ Bm, long long ldB, long long bOuter, long long bInner,
                      const float* __restrict__ bias,
                      CT* __restrict__ C, long long ldC, long long cOuter, long long cInner,
                      int Kdim, int HH, float alpha)
{
    constexpr bool C_IS_F32 = (sizeof(CT) == 4);

    const int z = blockIdx.z;
    const bf16_t* Ab = A  + (size_t)(z / HH) * (size_t)aOuter + (size_t)(z % HH) * (size_t)aInner;
    const bf16_t* Bb = Bm + (size_t)(z / HH) * (size_t)bOuter + (size_t)(z % HH) * (size_t)bInner;
    CT*           Cb = C  + (size_t)(z / HH) * (size_t)cOuter + (size_t)(z % HH) * (size_t)cInner;

    const int lane = threadIdx.x & 31;
    const int wave = threadIdx.x >> 5;
    const int wm   = wave / WN_WAVES;
    const int wn   = wave % WN_WAVES;
    const int r    = lane & 15;   // row/col within 16
    const int kh   = lane >> 4;   // K-half held by this lane

    const int m0 = blockIdx.y * (WM_WAVES * 64) + wm * 64;
    const int n0 = blockIdx.x * (WN_WAVES * 32) + wn * 32;

    v8f acc[4][2] = {};

    for (int k0 = 0; k0 < Kdim; k0 += 32) {
        // A fragments: 16x32, lane holds row (lane&15), 16 contiguous K at k0 + kh*16
        v16bf afrag[4];
#pragma unroll
        for (int i = 0; i < 4; ++i) {
            const bf16_t* p = Ab + (size_t)(m0 + i * 16 + r) * (size_t)ldA + k0 + kh * 16;
            v8bf lo = *(const v8bf*)p;
            v8bf hi = *(const v8bf*)(p + 8);
#pragma unroll
            for (int e = 0; e < 8; ++e) { afrag[i][e] = lo[e]; afrag[i][8 + e] = hi[e]; }
        }
        // B fragments: 32x16, lane holds column, 16 contiguous K at k0 + kh*16
        v16bf bfrag[2];
#pragma unroll
        for (int j = 0; j < 2; ++j) {
            const bf16_t* p = Bb + (size_t)(n0 + j * 16 + r) * (size_t)ldB + k0 + kh * 16;
            bfrag[j] = *(const v16bf*)p;
        }
#pragma unroll
        for (int i = 0; i < 4; ++i)
#pragma unroll
            for (int j = 0; j < 2; ++j)
                acc[i][j] = __builtin_amdgcn_wmma_f32_16x16x32_bf16(
                    false, afrag[i], false, bfrag[j], (short)0, acc[i][j], false, false);
    }

    // epilogue: C f32 16x16 layout -> m = base + 8*kh + vreg, n = lane&15
#pragma unroll
    for (int j = 0; j < 2; ++j) {
        const int n  = n0 + j * 16 + r;
        const float bvv = bias ? bias[n] : 0.0f;
#pragma unroll
        for (int i = 0; i < 4; ++i) {
#pragma unroll
            for (int vv = 0; vv < 8; ++vv) {
                const int m = m0 + i * 16 + kh * 8 + vv;
                const float val = (acc[i][j][vv] + bvv) * alpha;
                const size_t idx = TRANSC ? ((size_t)n * (size_t)ldC + m)
                                          : ((size_t)m * (size_t)ldC + n);
                if constexpr (C_IS_F32)
                    __builtin_nontemporal_store(val, &Cb[idx]);   // streaming fp32 out
                else
                    Cb[idx] = (CT)val;                            // L2-resident bf16
            }
        }
    }
}

// ---------------------------------------------------------------------------
// Fused softmax + P@V:
//   One WG (256 thr, 8 waves) owns 16 probs rows of one (b,h):
//   phase 1: NT-load 16x1024 fp32 scores, rowwise softmax (16-lane shuffle
//            groups), NT-store normalized fp32 probs (final output), keep a
//            bf16 copy in LDS (row stride padded to 1032 to skew banks).
//   phase 2: attn[16 x 64] = P(16x1024, LDS) @ V_h(1024x64, Vt global).
//            waves: j = wave&3 -> 16-col tile, wave>>2 -> K half (512).
//            Cross-half reduction through LDS, bf16 store to attn buffer.
// ---------------------------------------------------------------------------
__global__ __launch_bounds__(256)
void softmax_pv_kernel(float* __restrict__ probs,          // [B,H,S,S] fp32 (scores in, probs out)
                       const bf16_t* __restrict__ Vt,      // [B,D,S] bf16 (transposed V)
                       bf16_t* __restrict__ attn)          // [B,S,D] bf16
{
    constexpr int PSTR = 1032;                             // padded LDS row stride
    __shared__ __attribute__((aligned(16))) bf16_t Psh[16][PSTR];
    __shared__ float red[4][32][8];

    const int rb = blockIdx.x;                             // 16-row block within S
    const int z  = blockIdx.y;                             // b*H + h
    const int b  = z >> 4;
    const int h  = z & 15;

    float* prow = probs + (size_t)z * S_LEN * S_LEN + (size_t)rb * 16 * S_LEN;

    // ---------------- phase 1: softmax on 16 rows ----------------
    {
        const int row = threadIdx.x >> 4;                  // 0..15
        const int seg = threadIdx.x & 15;                  // 64-elem segment
        float* px = prow + (size_t)row * S_LEN + seg * 64;

        v4f x[16];
#pragma unroll
        for (int i = 0; i < 16; ++i)
            x[i] = __builtin_nontemporal_load((const v4f*)px + i);

        float mx = x[0][0];
#pragma unroll
        for (int i = 0; i < 16; ++i)
#pragma unroll
            for (int c = 0; c < 4; ++c) mx = fmaxf(mx, x[i][c]);
#pragma unroll
        for (int msk = 8; msk >= 1; msk >>= 1)
            mx = fmaxf(mx, __shfl_xor(mx, msk, 16));       // reduce across the row's 16 lanes

        float s = 0.0f;
#pragma unroll
        for (int i = 0; i < 16; ++i)
#pragma unroll
            for (int c = 0; c < 4; ++c) { x[i][c] = __expf(x[i][c] - mx); s += x[i][c]; }
#pragma unroll
        for (int msk = 8; msk >= 1; msk >>= 1)
            s += __shfl_xor(s, msk, 16);

        const float inv = 1.0f / s;
        bf16_t* lrow = &Psh[row][seg * 64];
#pragma unroll
        for (int i = 0; i < 16; ++i) {
            v4f e;
#pragma unroll
            for (int c = 0; c < 4; ++c) e[c] = x[i][c] * inv;
            __builtin_nontemporal_store(e, (v4f*)px + i);  // final fp32 probs
            v4bf o;
#pragma unroll
            for (int c = 0; c < 4; ++c) o[c] = (bf16_t)e[c];
            ((v4bf*)lrow)[i] = o;                          // bf16 copy for P@V
        }
    }
    __syncthreads();

    // ---------------- phase 2: attn = P @ V_h ----------------
    const int lane = threadIdx.x & 31;
    const int wave = threadIdx.x >> 5;
    const int r    = lane & 15;
    const int kh   = lane >> 4;
    const int j    = wave & 3;                             // 16-col tile of 64
    const int kbeg = (wave >> 2) * 512;                    // K half

    const bf16_t* Vb = Vt + (size_t)b * D_MODEL * S_LEN + (size_t)(h * D_KH) * S_LEN;

    v8f acc = {};
    for (int k0 = kbeg; k0 < kbeg + 512; k0 += 32) {
        // A fragment from LDS (row = r, 16 contiguous k at kh*16)
        const bf16_t* pa = &Psh[r][k0 + kh * 16];
        v8bf lo = *(const v8bf*)pa;
        v8bf hi = *(const v8bf*)(pa + 8);
        v16bf afrag;
#pragma unroll
        for (int e = 0; e < 8; ++e) { afrag[e] = lo[e]; afrag[8 + e] = hi[e]; }
        // B fragment from Vt: col d = j*16 + r, 16 contiguous k
        const bf16_t* pb = Vb + (size_t)(j * 16 + r) * S_LEN + k0 + kh * 16;
        v16bf bfrag = *(const v16bf*)pb;

        acc = __builtin_amdgcn_wmma_f32_16x16x32_bf16(
            false, afrag, false, bfrag, (short)0, acc, false, false);
    }

    if (wave >= 4) {
#pragma unroll
        for (int vv = 0; vv < 8; ++vv) red[j][lane][vv] = acc[vv];
    }
    __syncthreads();
    if (wave < 4) {
        bf16_t* ab = attn + (size_t)b * S_LEN * D_MODEL + (size_t)(rb * 16) * D_MODEL
                   + h * D_KH + j * 16 + r;
#pragma unroll
        for (int vv = 0; vv < 8; ++vv) {
            const int m = kh * 8 + vv;
            ab[(size_t)m * D_MODEL] = (bf16_t)(acc[vv] + red[j][lane][vv]);
        }
    }
}

// ---------------------------------------------------------------------------
extern "C" void kernel_launch(void* const* d_in, const int* in_sizes, int n_in,
                              void* d_out, int out_size, void* d_ws, size_t ws_size,
                              hipStream_t stream)
{
    const float* q  = (const float*)d_in[0];
    const float* k  = (const float*)d_in[1];
    const float* v  = (const float*)d_in[2];
    const float* Wq = (const float*)d_in[3];
    const float* bq = (const float*)d_in[4];
    const float* Wk = (const float*)d_in[5];
    const float* bk = (const float*)d_in[6];
    const float* Wv = (const float*)d_in[7];
    const float* bv = (const float*)d_in[8];
    const float* Wo = (const float*)d_in[9];
    const float* bo = (const float*)d_in[10];

    const size_t BSD = (size_t)B_SZ * S_LEN * D_MODEL;   // 8,388,608
    const size_t DD  = (size_t)D_MODEL * D_MODEL;        // 1,048,576
    const size_t SS  = (size_t)S_LEN * S_LEN;            // 1,048,576

    float* out   = (float*)d_out;
    float* probs = out + BSD;                            // [B,H,S,S] fp32

    // workspace (bf16): X (activations, later reused for attn), W, Q, K, Vt
    bf16_t* bufX = (bf16_t*)d_ws;        // BSD
    bf16_t* bufW = bufX + BSD;           // DD
    bf16_t* bufQ = bufW + DD;            // BSD (Q, pre-scaled by 1/8)
    bf16_t* bufK = bufQ + BSD;           // BSD
    bf16_t* bufV = bufK + BSD;           // BSD, transposed: Vt[b][d][s]

    const dim3 blk(256);
    const dim3 gProj(D_MODEL / 128, (B_SZ * S_LEN) / 128, 1);

    // ---- Q projection:  Qs = (x @ Wq^T + bq) * 0.125  (bf16) ----
    cvt_f32_bf16_kernel<<<dim3((unsigned)(BSD / 1024)), blk, 0, stream>>>(q, bufX);
    cvt_f32_bf16_kernel<<<dim3((unsigned)(DD / 1024)),  blk, 0, stream>>>(Wq, bufW);
    wmma_gemm_kernel<bf16_t, false, 2, 4><<<gProj, blk, 0, stream>>>(
        bufX, D_MODEL, 0, 0, bufW, D_MODEL, 0, 0, bq,
        bufQ, D_MODEL, 0, 0, D_MODEL, 1, 0.125f);

    // ---- K projection ----
    cvt_f32_bf16_kernel<<<dim3((unsigned)(BSD / 1024)), blk, 0, stream>>>(k, bufX);
    cvt_f32_bf16_kernel<<<dim3((unsigned)(DD / 1024)),  blk, 0, stream>>>(Wk, bufW);
    wmma_gemm_kernel<bf16_t, false, 2, 4><<<gProj, blk, 0, stream>>>(
        bufX, D_MODEL, 0, 0, bufW, D_MODEL, 0, 0, bk,
        bufK, D_MODEL, 0, 0, D_MODEL, 1, 1.0f);

    // ---- V projection, stored transposed per batch: Vt[b][d][s] ----
    cvt_f32_bf16_kernel<<<dim3((unsigned)(BSD / 1024)), blk, 0, stream>>>(v, bufX);
    cvt_f32_bf16_kernel<<<dim3((unsigned)(DD / 1024)),  blk, 0, stream>>>(Wv, bufW);
    wmma_gemm_kernel<bf16_t, true, 2, 4><<<dim3(D_MODEL / 128, S_LEN / 128, B_SZ), blk, 0, stream>>>(
        bufX, D_MODEL, (long long)(S_LEN * (size_t)D_MODEL), 0,
        bufW, D_MODEL, 0, 0, bv,
        bufV, S_LEN, (long long)(D_MODEL * (size_t)S_LEN), 0,
        D_MODEL, 1, 1.0f);

    // ---- scores[b,h] = Qs_h @ K_h^T  (fp32, NT-stored into probs region) ----
    wmma_gemm_kernel<float, false, 2, 4><<<dim3(S_LEN / 128, S_LEN / 128, B_SZ * N_HEADS), blk, 0, stream>>>(
        bufQ, D_MODEL, (long long)(S_LEN * (size_t)D_MODEL), D_KH,
        bufK, D_MODEL, (long long)(S_LEN * (size_t)D_MODEL), D_KH,
        nullptr,
        probs, S_LEN, (long long)(N_HEADS * SS), (long long)SS,
        D_KH, N_HEADS, 1.0f);

    // ---- fused softmax + P@V  (attn bf16 into bufX, reused) ----
    softmax_pv_kernel<<<dim3(S_LEN / 16, B_SZ * N_HEADS), blk, 0, stream>>>(
        probs, bufV, bufX);

    // ---- output projection: out = attn @ Wo^T + bo  (fp32, NT store) ----
    cvt_f32_bf16_kernel<<<dim3((unsigned)(DD / 1024)), blk, 0, stream>>>(Wo, bufW);
    wmma_gemm_kernel<float, false, 2, 4><<<gProj, blk, 0, stream>>>(
        bufX, D_MODEL, 0, 0, bufW, D_MODEL, 0, 0, bo,
        out, D_MODEL, 0, 0, D_MODEL, 1, 1.0f);
}
